// BitConvBlock_77764677861389
// MI455X (gfx1250) — matrix operand order repaired and checked by reading, not compiled, not run on hardware
//
#include <hip/hip_runtime.h>
#include <hip/hip_bf16.h>
#include <math.h>

typedef __attribute__((ext_vector_type(16))) _Float16 v16h;
typedef __attribute__((ext_vector_type(8)))  _Float16 v8h;
typedef __attribute__((ext_vector_type(4)))  _Float16 v4h;
typedef __attribute__((ext_vector_type(8)))  float    v8f;
typedef __attribute__((ext_vector_type(4)))  int      v4i;

#define CIN   512
#define COUT  512
#define KW    7
#define LEN   8192
#define BATCH 16
#define CKK   (CIN*KW)                 // 3584
#define WNUM  (COUT*CIN*KW)            // 1835008
#define NSTEP 112                      // 7 taps * 16 cin-chunks of 32

// workspace float layout
#define WS_PART0   0                   // [0,256): |W| partials
#define WS_S       256                 // s = mean(|W|)
#define WS_YPART   512                 // [512, 512+4096*2): per-block (sum, sumsq)
#define WS_STATS   9000                // [9000,9016) mean[b], [9016,9032) rstd[b]
#define WS_WQ_BYTE 40960               // f16 ternary weights start (16B aligned)

#define LDSTR 40                       // LDS row stride in halves (80B, 16B-aligned)

// gfx1250 async global->LDS copy (bypasses VGPRs, ASYNCcnt-tracked)
#if defined(__has_builtin)
#if __has_builtin(__builtin_amdgcn_global_load_async_to_lds_b128)
#define USE_ASYNC_LDS 1
// expected param types (from clang diagnostic): (int4 as1*, int4 as3*, imm i32, imm i32)
#define ASYNC_GPTR(p) ((__attribute__((address_space(1))) v4i*)(p))
#define ASYNC_LPTR(p) ((__attribute__((address_space(3))) v4i*)(p))
#endif
#endif

// ---------------- s = mean(|W|) ----------------
__global__ __launch_bounds__(256) void k_sumabs(const float* __restrict__ W, float* __restrict__ wsf) {
    __shared__ float red[256];
    float s = 0.f;
    for (int i = blockIdx.x * 256 + threadIdx.x; i < WNUM; i += 256 * 256)
        s += fabsf(W[i]);
    red[threadIdx.x] = s;
    __syncthreads();
    for (int off = 128; off; off >>= 1) {
        if (threadIdx.x < off) red[threadIdx.x] += red[threadIdx.x + off];
        __syncthreads();
    }
    if (threadIdx.x == 0) wsf[WS_PART0 + blockIdx.x] = red[0];
}

__global__ __launch_bounds__(256) void k_sumabs_reduce(float* __restrict__ wsf) {
    __shared__ float red[256];
    red[threadIdx.x] = wsf[WS_PART0 + threadIdx.x];
    __syncthreads();
    for (int off = 128; off; off >>= 1) {
        if (threadIdx.x < off) red[threadIdx.x] += red[threadIdx.x + off];
        __syncthreads();
    }
    if (threadIdx.x == 0) wsf[WS_S] = red[0] / (float)WNUM;   // store s directly
}

// ---------------- ternary quantize: W[co][ci][k] -> wq[co][k][ci] in {-1,0,1} f16 ----------------
__global__ __launch_bounds__(256) void k_quant(const float* __restrict__ W, const float* __restrict__ wsf,
                                               _Float16* __restrict__ wq) {
    const int i = blockIdx.x * 256 + threadIdx.x;
    if (i >= WNUM) return;
    const float s = wsf[WS_S];
    float q = rintf(W[i] / (s + 1e-5f));          // jnp.round = round-half-even
    q = fminf(1.f, fmaxf(-1.f, q));
    const int co  = i / CKK;
    const int rem = i - co * CKK;
    const int ci  = rem / KW;
    const int k   = rem - ci * KW;
    wq[(size_t)co * CKK + k * CIN + ci] = (_Float16)q;
}

// ---------------- implicit-GEMM conv via WMMA f16, double-buffered LDS ----------------
// grid: (L/128=64, COUT/128=4, B=16), block 256 (8 waves; wave -> 64x32 sub-tile)
__global__ __launch_bounds__(256) void k_gemm(const float* __restrict__ x, const _Float16* __restrict__ wq,
                                              const float* __restrict__ wsf, float* __restrict__ out,
                                              float* __restrict__ ypart) {
    __shared__ _Float16 lA[2][128 * LDSTR];   // A: 128 Cout x 32 Cin halves, ping/pong
    __shared__ _Float16 lB[2][128 * LDSTR];   // B: [n][k] (x transposed, K-contiguous), ping/pong
    __shared__ float    red[512];

    const int t    = threadIdx.x;
    const int lane = t & 31;
    const int wave = t >> 5;
    const int row  = lane & 15;             // M row / N col within a 16x16 tile
    const int kg   = lane >> 4;             // lane group
    const int wm   = wave & 1;              // 64-row slab
    const int wn   = wave >> 1;             // 32-col slab

    const int b     = blockIdx.z;
    const int cout0 = blockIdx.y * 128;
    const int l0    = blockIdx.x * 128;
    const float* xb = x + (size_t)b * CIN * LEN;

    const v8f vzero = {};
    v8f acc[8];
#pragma unroll
    for (int i = 0; i < 8; ++i) acc[i] = vzero;

    // loader roles
    const int rA = t >> 1;              // A: row 0..127
    const int oA = (t & 1) * 16;        // A: half-offset 0 or 16
    const int nB = t & 127;             // B: n (L) index 0..127
    const int kh = t >> 7;              // B: k-half 0 or 1
    const _Float16* wqT = wq + (size_t)(cout0 + rA) * CKK + oA;   // + tap*CIN + cin0

    // ---- staging helpers ----
#if !defined(USE_ASYNC_LDS)
    v8h aLo = {}, aHi = {};
#endif
    float xr[16];
    float msk = 0.f;

    auto issue_stage = [&](int step, int buf) {       // global-side work for `step`
        const int tap  = step >> 4;
        const int cin0 = (step & 15) << 5;
        // A tile: pure f16 copy
        {
            const _Float16* src = wqT + tap * CIN + cin0;
#if defined(USE_ASYNC_LDS)
            __builtin_amdgcn_global_load_async_to_lds_b128(
                ASYNC_GPTR(src), ASYNC_LPTR(&lA[buf][rA * LDSTR + oA]), 0, 0);
            __builtin_amdgcn_global_load_async_to_lds_b128(
                ASYNC_GPTR(src + 8), ASYNC_LPTR(&lA[buf][rA * LDSTR + oA + 8]), 0, 0);
#else
            aLo = *(const v8h*)src;
            aHi = *(const v8h*)(src + 8);
#endif
        }
        // B tile: f32 loads, branchless pad mask
        {
            const int l  = l0 + nB + tap - 3;
            const int lc = l < 0 ? 0 : (l >= LEN ? LEN - 1 : l);
            msk = ((unsigned)l < (unsigned)LEN) ? 1.f : 0.f;
            const float* xs = xb + (size_t)(cin0 + kh * 16) * LEN + lc;
#pragma unroll
            for (int j = 0; j < 16; ++j) xr[j] = xs[(size_t)j * LEN];
        }
    };

    auto store_stage = [&](int buf) {                 // LDS-side work
#if !defined(USE_ASYNC_LDS)
        *(v8h*)&lA[buf][rA * LDSTR + oA]     = aLo;
        *(v8h*)&lA[buf][rA * LDSTR + oA + 8] = aHi;
#endif
#pragma unroll
        for (int j4 = 0; j4 < 4; ++j4) {
            v4h p;
#pragma unroll
            for (int e = 0; e < 4; ++e) p[e] = (_Float16)(xr[j4 * 4 + e] * msk);
            *(v4h*)&lB[buf][nB * LDSTR + kh * 16 + j4 * 4] = p;
        }
    };

    // ---- prologue: stage step 0 into buffer 0 ----
    issue_stage(0, 0);
    store_stage(0);
#if defined(USE_ASYNC_LDS)
    asm volatile("s_wait_asynccnt 0x0" ::: "memory");
#endif
    __syncthreads();

    // ---- main pipeline: one barrier per step ----
    for (int step = 0; step < NSTEP; ++step) {
        const int cur = step & 1, nxt = cur ^ 1;
        const bool more = (step + 1) < NSTEP;
        if (more) issue_stage(step + 1, nxt);         // global loads overlap compute below

        // fragments (ISA 7.12.2: 16-bit A 16x32 layout; B mirrored K-striping)
        v16h af[4], bf[2];
#pragma unroll
        for (int mt = 0; mt < 4; ++mt) {
            const int base = (wm * 64 + mt * 16 + row) * LDSTR + kg * 8;
            v8h lo = *(const v8h*)&lA[cur][base];
            v8h hi = *(const v8h*)&lA[cur][base + 16];
            af[mt] = __builtin_shufflevector(lo, hi, 0,1,2,3,4,5,6,7,8,9,10,11,12,13,14,15);
        }
#pragma unroll
        for (int nt = 0; nt < 2; ++nt) {
            const int base = (wn * 32 + nt * 16 + row) * LDSTR + kg * 8;
            v8h lo = *(const v8h*)&lB[cur][base];
            v8h hi = *(const v8h*)&lB[cur][base + 16];
            bf[nt] = __builtin_shufflevector(lo, hi, 0,1,2,3,4,5,6,7,8,9,10,11,12,13,14,15);
        }
#pragma unroll
        for (int mt = 0; mt < 4; ++mt)
#pragma unroll
            for (int nt = 0; nt < 2; ++nt)
                acc[mt * 2 + nt] = __builtin_amdgcn_wmma_f32_16x16x32_f16(
                    false, af[mt], false, bf[nt], (short)0, acc[mt * 2 + nt], false, false);

        if (more) store_stage(nxt);                   // write next stage (other buffer)
#if defined(USE_ASYNC_LDS)
        asm volatile("s_wait_asynccnt 0x0" ::: "memory");
#endif
        __syncthreads();   // publishes nxt, and frees cur for overwrite next iteration
    }

    // ---- epilogue: scale by s, store y, accumulate block stats ----
    const float s = wsf[WS_S];
    float ps = 0.f, pss = 0.f;
#pragma unroll
    for (int mt = 0; mt < 4; ++mt) {
#pragma unroll
        for (int nt = 0; nt < 2; ++nt) {
            const int l = l0 + wn * 32 + nt * 16 + row;
#pragma unroll
            for (int r = 0; r < 8; ++r) {
                const int m = cout0 + wm * 64 + mt * 16 + r + 8 * kg;   // C/D: M=r(+8), N=lane&15
                const float v = acc[mt * 2 + nt][r] * s;
                out[((size_t)b * COUT + m) * LEN + l] = v;
                ps  += v;
                pss += v * v;
            }
        }
    }
    red[t] = ps; red[256 + t] = pss;
    __syncthreads();
    for (int off = 128; off; off >>= 1) {
        if (t < off) { red[t] += red[t + off]; red[256 + t] += red[256 + t + off]; }
        __syncthreads();
    }
    if (t == 0) {
        const int pb = blockIdx.y * 64 + blockIdx.x;        // 0..255 within batch
        float* p = ypart + ((size_t)b * 256 + pb) * 2;
        p[0] = red[0]; p[1] = red[256];
    }
}

// ---------------- per-batch mean / rstd ----------------
__global__ __launch_bounds__(256) void k_stats(float* __restrict__ wsf) {
    __shared__ float red[512];
    const int b = blockIdx.x, t = threadIdx.x;
    const float* p = wsf + WS_YPART + ((size_t)b * 256 + t) * 2;
    red[t] = p[0]; red[256 + t] = p[1];
    __syncthreads();
    for (int off = 128; off; off >>= 1) {
        if (t < off) { red[t] += red[t + off]; red[256 + t] += red[256 + t + off]; }
        __syncthreads();
    }
    if (t == 0) {
        const float inv = 1.f / (float)((size_t)COUT * LEN);
        const float mu  = red[0] * inv;
        const float var = red[256] * inv - mu * mu;
        wsf[WS_STATS + b]      = mu;
        wsf[WS_STATS + 16 + b] = rsqrtf(var + 1e-5f);
    }
}

// ---------------- norm + affine + snake, in-place, float4 ----------------
__global__ __launch_bounds__(256) void k_final(float* __restrict__ out, const float* __restrict__ wsf,
                                               const float* __restrict__ gnw, const float* __restrict__ gnb,
                                               const float* __restrict__ alpha, const float* __restrict__ phase) {
    const size_t idx4 = ((size_t)blockIdx.x * 256 + threadIdx.x) * 4;
    const int b = (int)(idx4 >> 22);            // 512*8192 = 2^22 per batch
    const int c = (int)((idx4 >> 13) & 511);    // L = 2^13
    const float mu = wsf[WS_STATS + b];
    const float rs = wsf[WS_STATS + 16 + b];
    const float gw = gnw[c] * rs;
    const float gb = gnb[c] - mu * gw;
    const float a  = alpha[c];
    const float ph = phase[c];
    const float inva = 1.f / (a + 1e-9f);

    float4 v = *reinterpret_cast<const float4*>(out + idx4);
    float ys[4] = {v.x, v.y, v.z, v.w};
#pragma unroll
    for (int j = 0; j < 4; ++j) {
        const float yn = fmaf(ys[j], gw, gb);
        const float sn = sinf(fmaf(a, yn, ph));
        ys[j] = fmaf(sn * sn, inva, yn);
    }
    v.x = ys[0]; v.y = ys[1]; v.z = ys[2]; v.w = ys[3];
    *reinterpret_cast<float4*>(out + idx4) = v;
}

extern "C" void kernel_launch(void* const* d_in, const int* in_sizes, int n_in,
                              void* d_out, int out_size, void* d_ws, size_t ws_size,
                              hipStream_t stream) {
    const float* x     = (const float*)d_in[0];
    const float* W     = (const float*)d_in[1];
    const float* gnw   = (const float*)d_in[2];
    const float* gnb   = (const float*)d_in[3];
    const float* alpha = (const float*)d_in[4];
    const float* phase = (const float*)d_in[5];
    float*  out  = (float*)d_out;
    float*  wsf  = (float*)d_ws;
    float*  ypart = wsf + WS_YPART;
    _Float16* wq = (_Float16*)((char*)d_ws + WS_WQ_BYTE);

    k_sumabs<<<256, 256, 0, stream>>>(W, wsf);
    k_sumabs_reduce<<<1, 256, 0, stream>>>(wsf);
    k_quant<<<(WNUM + 255) / 256, 256, 0, stream>>>(W, wsf, wq);

    dim3 g(LEN / 128, COUT / 128, BATCH);          // (64, 4, 16)
    k_gemm<<<g, 256, 0, stream>>>(x, wq, wsf, out, ypart);

    k_stats<<<BATCH, 256, 0, stream>>>(wsf);

    const size_t total = (size_t)BATCH * COUT * LEN;  // 67108864
    k_final<<<(unsigned)(total / (256 * 4)), 256, 0, stream>>>(out, wsf, gnw, gnb, alpha, phase);
}
